// TaskAttention_50165218017857
// MI455X (gfx1250) — matrix operand order repaired and checked by reading, not compiled, not run on hardware
//
#include <hip/hip_runtime.h>
#include <hip/hip_bf16.h>
#include <math.h>

typedef __attribute__((ext_vector_type(16))) _Float16 v16h;
typedef __attribute__((ext_vector_type(8)))  float    v8f;

constexpr int S = 4096;
constexpr int B = 32;
constexpr int D = 512;
constexpr int NSEL = S / 2;   // 2048 masked positions per row

// ---------------------------------------------------------------------------
// Pass 1: scores w[b,s] = sum_d x[s,b,d] * te[b,d]
// One wave computes a 16-row s-tile for one b using v_wmma_f32_16x16x32_f16:
//   A[i][k] = (f16) x[s0+i, b, d0+k]      (16x32 tile, f32->f16 cvt in regs)
//   B[k][j] = (f16) te[b, d0+k]           (broadcast over the 16 columns)
//   C accumulates in f32; every column of C holds the same dot products.
// te row is staged into LDS with ASYNC global->LDS copies (CDNA5 async path),
// converted once to f16 so each lane reads its B fragment as 2x ds_load_b128.
// The kernel is HBM-bound (x streamed once: 256 MB @ 23.3 TB/s ~ 11 us).
// ---------------------------------------------------------------------------
__global__ __launch_bounds__(256)
void scores_wmma_kernel(const float* __restrict__ x,
                        const float* __restrict__ te,
                        float* __restrict__ w) {
  __shared__ float    te_stage[D];   // raw f32 row, filled by async copy
  __shared__ _Float16 te_h[D];       // converted once, feeds B fragments

  const int b = blockIdx.y;

  // --- async global->LDS staging of te row (2 KB), wave 0 only -------------
  if (threadIdx.x < 32) {
    const unsigned long long gbase =
        (unsigned long long)(const void*)(te + (size_t)b * D);
    const unsigned lds_base = (unsigned)(uintptr_t)(void*)te_stage;
#pragma unroll
    for (int it = 0; it < 4; ++it) {
      const unsigned byte_off = (unsigned)((it * 32 + threadIdx.x) * 16);
      const unsigned lofs = lds_base + byte_off;
      asm volatile("global_load_async_to_lds_b128 %0, %1, %2"
                   :: "v"(lofs), "v"(byte_off), "s"(gbase)
                   : "memory");
    }
    asm volatile("s_wait_asynccnt 0" ::: "memory");
  }
  __syncthreads();

  // one-time f32 -> f16 conversion of te row
  for (int i = threadIdx.x; i < D; i += 256)
    te_h[i] = (_Float16)te_stage[i];
  __syncthreads();

  const int wave = threadIdx.x >> 5;
  const int lane = threadIdx.x & 31;
  const int g    = lane >> 4;   // half-wave: 0 or 1
  const int m    = lane & 15;   // A row within tile (== C column id)
  const int s0   = (blockIdx.x * 8 + wave) * 16;

  const float* __restrict__ xrow = x + ((size_t)(s0 + m) * B + b) * D;

  v8f c = {};
#pragma unroll 4
  for (int d0 = 0; d0 < D; d0 += 32) {
    __builtin_prefetch(xrow + d0 + 256, 0, 3);  // global_prefetch_b8, WGP scope

    // A fragment (16-bit A 16x32 layout):
    //   lane(g,m): VGPR v   -> K = g*8 + 2v, 2v+1        (halves 0..7)
    //              VGPR 4+v -> K = 16 + g*8 + 2v, 2v+1   (halves 8..15)
    const float4* pa = reinterpret_cast<const float4*>(xrow + d0 + g * 8);
    const float4  f0 = pa[0];
    const float4  f1 = pa[1];
    const float4* pb = reinterpret_cast<const float4*>(xrow + d0 + 16 + g * 8);
    const float4  f2 = pb[0];
    const float4  f3 = pb[1];

    v16h a;
    a[0]  = (_Float16)f0.x;  a[1]  = (_Float16)f0.y;
    a[2]  = (_Float16)f0.z;  a[3]  = (_Float16)f0.w;
    a[4]  = (_Float16)f1.x;  a[5]  = (_Float16)f1.y;
    a[6]  = (_Float16)f1.z;  a[7]  = (_Float16)f1.w;
    a[8]  = (_Float16)f2.x;  a[9]  = (_Float16)f2.y;
    a[10] = (_Float16)f2.z;  a[11] = (_Float16)f2.w;
    a[12] = (_Float16)f3.x;  a[13] = (_Float16)f3.y;
    a[14] = (_Float16)f3.z;  a[15] = (_Float16)f3.w;

    // B fragment (16-bit B 32x16): lane holds K = g*8 + t (t<8), 16 + g*8 + (t-8) (t>=8);
    // te broadcast across columns -> two 16B LDS reads.
    v16h bb;
#pragma unroll
    for (int t = 0; t < 8; ++t) {
      bb[t]     = te_h[d0 + g * 8 + t];
      bb[8 + t] = te_h[d0 + 16 + g * 8 + t];
    }

    c = __builtin_amdgcn_wmma_f32_16x16x32_f16(
        /*neg_a=*/false, a, /*neg_b=*/false, bb,
        /*c_mod=*/(short)0, c, /*reuse_a=*/false, /*reuse_b=*/false);
  }

  // C/D layout: lane L, VGPR r -> M = r + 8*(L>>4), N = L&15.
  // Take column N==0 (lanes 0 and 16): together they cover M = 0..15.
  if (m == 0) {
#pragma unroll
    for (int r = 0; r < 8; ++r)
      w[(size_t)b * S + s0 + g * 8 + r] = c[r];
  }
}

// ---------------------------------------------------------------------------
// Pass 2: one block per batch row b.
//   key[s] = gumbel(b,s) - w[s]   (top-NSEL of key == sampled/masked set,
//   since the row constants in log(p_inv)+g do not change the top-k order)
//   Find NSEL-th largest key via 32-step bit search on order-preserving
//   uint mapping, then masked softmax of w over the unmasked positions.
// ---------------------------------------------------------------------------
__device__ __forceinline__ unsigned pcg_hash(unsigned v) {
  unsigned h = v * 747796405u + 2891336453u;
  h = ((h >> ((h >> 28u) + 4u)) ^ h) * 277803737u;
  return (h >> 22u) ^ h;
}

__global__ __launch_bounds__(256)
void mask_softmax_kernel(const float* __restrict__ w,
                         float* __restrict__ out) {
  __shared__ float    w_s[S];
  __shared__ unsigned k_s[S];
  __shared__ unsigned redu[256];
  __shared__ float    redf[256];

  const int b   = blockIdx.x;
  const int tid = threadIdx.x;

  // Load row, build Gumbel keys.
  for (int s = tid; s < S; s += 256) {
    float ws = w[(size_t)b * S + s];
    w_s[s] = ws;
    unsigned h = pcg_hash((unsigned)(b * S + s) ^ 0x9E3779B9u);
    float u = (float)(h >> 8) * (1.0f / 16777216.0f) + 1e-7f;
    float gmb = -logf(-logf(u));
    float key = gmb - ws;
    unsigned kb = __float_as_uint(key);
    kb = (kb & 0x80000000u) ? ~kb : (kb | 0x80000000u);  // monotone map
    k_s[s] = kb;
  }
  __syncthreads();

  // Bit-search the NSEL-th largest key: max T with count(key >= T) >= NSEL.
  unsigned T = 0u;
  for (int bit = 31; bit >= 0; --bit) {
    const unsigned cand = T | (1u << bit);
    unsigned cnt = 0;
    for (int s = tid; s < S; s += 256) cnt += (k_s[s] >= cand) ? 1u : 0u;
    redu[tid] = cnt;
    __syncthreads();
    for (int off = 128; off > 0; off >>= 1) {
      if (tid < off) redu[tid] += redu[tid + off];
      __syncthreads();
    }
    const unsigned total = redu[0];
    __syncthreads();
    if (total >= (unsigned)NSEL) T = cand;
  }
  // masked (sampled -> -inf) iff key >= T ; unmasked iff key < T.

  // Row max over unmasked.
  float mx = -3.4e38f;
  for (int s = tid; s < S; s += 256)
    if (k_s[s] < T) mx = fmaxf(mx, w_s[s]);
  redf[tid] = mx;
  __syncthreads();
  for (int off = 128; off > 0; off >>= 1) {
    if (tid < off) redf[tid] = fmaxf(redf[tid], redf[tid + off]);
    __syncthreads();
  }
  mx = redf[0];
  __syncthreads();

  // Exp-sum over unmasked.
  float acc = 0.0f;
  for (int s = tid; s < S; s += 256)
    if (k_s[s] < T) acc += expf(w_s[s] - mx);
  redf[tid] = acc;
  __syncthreads();
  for (int off = 128; off > 0; off >>= 1) {
    if (tid < off) redf[tid] += redf[tid + off];
    __syncthreads();
  }
  const float inv = 1.0f / redf[0];
  __syncthreads();

  // Write [S, B, 1] output.
  for (int s = tid; s < S; s += 256) {
    float v = (k_s[s] < T) ? expf(w_s[s] - mx) * inv : 0.0f;
    out[(size_t)s * B + b] = v;
  }
}

extern "C" void kernel_launch(void* const* d_in, const int* in_sizes, int n_in,
                              void* d_out, int out_size, void* d_ws, size_t ws_size,
                              hipStream_t stream) {
  (void)in_sizes; (void)n_in; (void)out_size; (void)ws_size;
  const float* x  = (const float*)d_in[0];  // [S, B, D]
  const float* te = (const float*)d_in[1];  // [B, D, 1]
  float* out = (float*)d_out;               // [S, B, 1]
  float* w   = (float*)d_ws;                // [B, S] scores scratch (512 KB)

  dim3 g1(S / (8 * 16), B);   // 8 waves/block, 16 s-rows/wave
  scores_wmma_kernel<<<g1, 256, 0, stream>>>(x, te, w);
  mask_softmax_kernel<<<B, 256, 0, stream>>>(w, out);
}
